// CrossAttention_867583394239
// MI455X (gfx1250) — matrix operand order repaired and checked by reading, not compiled, not run on hardware
//
#include <hip/hip_runtime.h>

#define HIDDEN   2048
#define HEADS    16
#define HEAD_DIM 128
#define BATCH    2
#define SEQQ     2048
#define SEQK     2048

typedef __bf16 bf16;
typedef __attribute__((ext_vector_type(16))) __bf16        v16bf;
typedef __attribute__((ext_vector_type(8)))  float         v8f;
typedef __attribute__((ext_vector_type(4)))  unsigned int  v4u;
typedef __attribute__((ext_vector_type(8)))  int           i32x8;
typedef __attribute__((ext_vector_type(4)))  int           i32x4;

#if __has_include(<hip/amd_detail/amd_gfx1250_TDM.h>)
#define TDM_SIX_ARGS 1
#else
#define TDM_SIX_ARGS 0
#endif

#if defined(__HIP_DEVICE_COMPILE__) && defined(__gfx1250__) && \
    __has_builtin(__builtin_amdgcn_tensor_load_to_lds)
#define USE_TDM 1
#else
#define USE_TDM 0
#endif

// ---------------- helpers ----------------

static __device__ __forceinline__ bf16 f2bf(float f) {
  unsigned u = __builtin_bit_cast(unsigned, f);
  unsigned r = (u + 0x7FFFu + ((u >> 16) & 1u)) >> 16;   // round-to-nearest-even
  unsigned short s = (unsigned short)r;
  return __builtin_bit_cast(bf16, s);
}
static __device__ __forceinline__ float bf2f(bf16 b) {
  unsigned short s = __builtin_bit_cast(unsigned short, b);
  unsigned u = ((unsigned)s) << 16;
  return __builtin_bit_cast(float, u);
}

struct Pair128 { v4u lo, hi; };

// 16 contiguous bf16 (two 16B loads) -> B-matrix fragment feed
static __device__ __forceinline__ v16bf ld16(const bf16* p) {
  Pair128 t;
  t.lo = *(const v4u*)p;
  t.hi = *(const v4u*)(p + 8);
  return __builtin_bit_cast(v16bf, t);
}
// A-matrix 16x32 bf16 fragment: lane holds K = kbase + 16*laneHi + {0..15}
static __device__ __forceinline__ v16bf ldA(const bf16* row, int kbase, int laneHi) {
  const bf16* p = row + kbase + 16 * laneHi;
  Pair128 t;
  t.lo = *(const v4u*)p;
  t.hi = *(const v4u*)(p + 8);
  return __builtin_bit_cast(v16bf, t);
}
static __device__ __forceinline__ v8f zero8() {
  v8f z;
  for (int i = 0; i < 8; ++i) z[i] = 0.0f;
  return z;
}
static __device__ __forceinline__ v8f wmma_bf16(v16bf a, v16bf b, v8f c) {
  return __builtin_amdgcn_wmma_f32_16x16x32_bf16(false, a, false, b, (short)0, c,
                                                 false, false);
}

static __device__ __forceinline__ unsigned lds_off(const void* p) {
  // flat->LDS mapping truncates to addr[31:0]; low bits carry the LDS offset
  return (unsigned)(unsigned long long)p;
}

#if USE_TDM
// Tensor Data Mover: 2D tile (tile_d0 elems/row x tile_d1 rows, bf16) from
// global (row stride stride0_elems) into LDS, with DMA-side row padding so the
// LDS layout lands at (tile_d0 + pad) element stride.
static __device__ __forceinline__ void tdm_load_2d(unsigned ldsoff,
                                                   const void* gptr,
                                                   unsigned tile_d0_elems,
                                                   unsigned tile_d1_rows,
                                                   unsigned stride0_elems,
                                                   unsigned pad_interval_code,
                                                   unsigned pad_amount_code) {
  unsigned long long ga = (unsigned long long)gptr;
  v4u g0;
  g0[0] = 1u;                                             // count=1 (valid, user)
  g0[1] = ldsoff;                                         // lds_addr
  g0[2] = (unsigned)ga;                                   // global_addr[31:0]
  g0[3] = ((unsigned)(ga >> 32) & 0x01FFFFFFu) | (2u << 30);  // [56:32] | type=2
  i32x8 g1;
  g1[0] = (int)((1u << 16) |                              // data_size = 2 bytes
                (1u << 20) |                              // pad_enable
                (pad_interval_code << 22) | (pad_amount_code << 25));
  unsigned td0 = stride0_elems;                           // tensor_dim0
  unsigned td1 = 0x40000000u;                             // tensor_dim1 (no OOB)
  g1[1] = (int)((td0 & 0xFFFFu) << 16);                   // barrier_addr=0 | td0.lo
  g1[2] = (int)(((td0 >> 16) & 0xFFFFu) | ((td1 & 0xFFFFu) << 16));
  g1[3] = (int)(((td1 >> 16) & 0xFFFFu) | (tile_d0_elems << 16));
  g1[4] = (int)(tile_d1_rows & 0xFFFFu);                  // tile_dim1 | tile_dim2=0
  g1[5] = (int)stride0_elems;                             // dim0_stride[31:0]
  g1[6] = 0;                                              // dim0_stride hi | dim1_stride lo
  g1[7] = 0;
  i32x4 z4 = {0, 0, 0, 0};
#if TDM_SIX_ARGS
  i32x8 z8 = {0, 0, 0, 0, 0, 0, 0, 0};
  __builtin_amdgcn_tensor_load_to_lds(g0, g1, z4, z4, z8, 0);
#else
  __builtin_amdgcn_tensor_load_to_lds(g0, g1, z4, z4, 0);
#endif
}
#endif

// ---------------- elementwise convert f32 -> bf16 ----------------

__global__ __launch_bounds__(256) void k_f32_to_bf16(const float* __restrict__ in,
                                                     bf16* __restrict__ out, int n4) {
  int i = blockIdx.x * blockDim.x + threadIdx.x;
  if (i >= n4) return;
  float4 f = ((const float4*)in)[i];
  unsigned long long pk =
      (unsigned long long)__builtin_bit_cast(unsigned short, f2bf(f.x)) |
      ((unsigned long long)__builtin_bit_cast(unsigned short, f2bf(f.y)) << 16) |
      ((unsigned long long)__builtin_bit_cast(unsigned short, f2bf(f.z)) << 32) |
      ((unsigned long long)__builtin_bit_cast(unsigned short, f2bf(f.w)) << 48);
  *(unsigned long long*)(out + 4ull * i) = pk;
}

// ---------------- RoPE (interleaved pairs, as reference) ----------------

__global__ __launch_bounds__(256) void k_rope(const bf16* __restrict__ in,
                                              bf16* __restrict__ out, int npairs) {
  int i = blockIdx.x * blockDim.x + threadIdx.x;
  if (i >= npairs) return;
  int f = i & 63;
  int h = (i >> 6) & 15;
  int s = (i >> 10) & 2047;
  int b = i >> 21;
  float inv = __expf(-((float)(2 * f) * (1.0f / 128.0f)) * 9.210340371976184f);
  float c, sn;
  __sincosf((float)s * inv, &c, &sn);
  long long base = (((long long)(b * 2048 + s)) * HIDDEN) + h * HEAD_DIM + 2 * f;
  float x0 = bf2f(in[base]);
  float x1 = bf2f(in[base + 1]);
  out[base]     = f2bf(x0 * c - x1 * sn);
  out[base + 1] = f2bf(x0 * sn + x1 * c);
}

// ---------------- GEMM  C[M,N] = A[M,K] * W[N,K]^T  (bf16 WMMA, TDM tiles) ------

template <bool OUT_F32>
__global__ __launch_bounds__(256) void k_gemm_nt(const bf16* __restrict__ A,
                                                 const bf16* __restrict__ W,
                                                 void* __restrict__ C,
                                                 int M, int N, int K) {
  const int LDT = 40;  // 32 + 8 pad (80B rows, 16B aligned; TDM pad 16DWx4DW)
  __shared__ __align__(16) bf16 As[2][128 * LDT];
  __shared__ __align__(16) bf16 Bs[2][128 * LDT];

  const int tid = threadIdx.x;
  const int lane = tid & 31, wave = tid >> 5;
  const int laneHi = lane >> 4, lane16 = lane & 15;
  const int m0 = blockIdx.y * 128, n0 = blockIdx.x * 128;
  const int wm = (wave >> 1) * 32;
  const int wn = (wave & 1) * 64;

  v8f acc[2][4];
  for (int mi = 0; mi < 2; ++mi)
    for (int ni = 0; ni < 4; ++ni) acc[mi][ni] = zero8();

  const int nk = K / 32;

  auto fill = [&](int buf, int kt) {
    const int kb = kt * 32;
#if USE_TDM
    if (tid < 32) {   // one wave issues both DMA descriptors
      tdm_load_2d(lds_off(&As[buf][0]), &A[(long long)m0 * K + kb], 32, 128,
                  (unsigned)K, /*interval 16DW*/ 3, /*pad 4DW*/ 3);
      tdm_load_2d(lds_off(&Bs[buf][0]), &W[(long long)n0 * K + kb], 32, 128,
                  (unsigned)K, 3, 3);
    }
#else
#pragma unroll
    for (int j = 0; j < 2; ++j) {
      int c  = tid + 256 * j;
      int r  = c >> 2;
      int cb = (c & 3) * 8;
      *(v4u*)&As[buf][r * LDT + cb] = *(const v4u*)&A[(long long)(m0 + r) * K + kb + cb];
      *(v4u*)&Bs[buf][r * LDT + cb] = *(const v4u*)&W[(long long)(n0 + r) * K + kb + cb];
    }
#endif
  };
  auto wait_fill = [&]() {
#if USE_TDM
    if (tid < 32) __builtin_amdgcn_s_wait_tensorcnt(0);
#endif
  };

  fill(0, 0);
  wait_fill();
  __syncthreads();

  for (int kt = 0; kt < nk; ++kt) {
    const int cur = kt & 1;
    if (kt + 1 < nk) fill(cur ^ 1, kt + 1);   // DMA next tile while computing

    const bf16* Asb = As[cur];
    const bf16* Bsb = Bs[cur];
    v16bf bfr[4];
#pragma unroll
    for (int ni = 0; ni < 4; ++ni)
      bfr[ni] = ld16(&Bsb[(wn + ni * 16 + lane16) * LDT + 16 * laneHi]);
#pragma unroll
    for (int mi = 0; mi < 2; ++mi) {
      v16bf a = ldA(&Asb[(wm + mi * 16 + lane16) * LDT], 0, laneHi);
#pragma unroll
      for (int ni = 0; ni < 4; ++ni)
        acc[mi][ni] = wmma_bf16(a, bfr[ni], acc[mi][ni]);
    }

    __syncthreads();           // all waves done reading cur
    if (kt + 1 < nk) wait_fill();
    __syncthreads();           // next buffer visible to all
  }

#pragma unroll
  for (int mi = 0; mi < 2; ++mi)
#pragma unroll
    for (int ni = 0; ni < 4; ++ni) {
      int rm = m0 + wm + mi * 16 + laneHi * 8;
      int cn = n0 + wn + ni * 16 + lane16;
#pragma unroll
      for (int r = 0; r < 8; ++r) {
        float v = acc[mi][ni][r];
        if (OUT_F32) ((float*)C)[(long long)(rm + r) * N + cn] = v;
        else         ((bf16*)C)[(long long)(rm + r) * N + cn]  = f2bf(v);
      }
    }
}

// ---------------- Flash cross-attention (bf16 WMMA, online softmax) -------------

__global__ __launch_bounds__(256) void k_attn(const bf16* __restrict__ Q,
                                              const bf16* __restrict__ Kg,
                                              const bf16* __restrict__ Vg,
                                              const int*  __restrict__ maskg,
                                              bf16* __restrict__ O) {
  const int LQ = 136, LK = 136, LV = 72, LP = 72;
  extern __shared__ __align__(16) char smem[];
  bf16* Qs  = (bf16*)smem;          // [128][136]
  bf16* Ks  = Qs + 128 * LQ;        // [64][136]
  bf16* Vst = Ks + 64 * LK;         // [128][72] transposed V
  bf16* Ps  = Vst + 128 * LV;       // [8][16][72] per-wave P scratch

  const int tid = threadIdx.x;
  const int lane = tid & 31, wave = tid >> 5;
  const int laneHi = lane >> 4, lane16 = lane & 15;
  const int q0 = blockIdx.x * 128;
  const int h  = blockIdx.y;
  const int b  = blockIdx.z;
  const float scale = 0.08838834764831845f;  // 1/sqrt(128)

  // Q tile: TDM DMA (pad 64DW interval x 4DW -> stride 136) or manual
#if USE_TDM
  if (tid < 32) {
    tdm_load_2d(lds_off(Qs),
                &Q[((long long)(b * SEQQ + q0)) * HIDDEN + h * HEAD_DIM],
                128, 128, HIDDEN, /*64DW*/ 5, /*4DW*/ 3);
    __builtin_amdgcn_s_wait_tensorcnt(0);
  }
#else
#pragma unroll
  for (int j = 0; j < 8; ++j) {
    int c = tid + 256 * j;
    int r = c >> 4;
    int cb = (c & 15) * 8;
    *(v4u*)&Qs[r * LQ + cb] =
        *(const v4u*)&Q[((long long)(b * SEQQ + q0 + r)) * HIDDEN + h * HEAD_DIM + cb];
  }
#endif

  v8f Oacc[8];
#pragma unroll
  for (int nd = 0; nd < 8; ++nd) Oacc[nd] = zero8();
  float mrun[8], lrun[8];
#pragma unroll
  for (int r = 0; r < 8; ++r) { mrun[r] = -1e30f; lrun[r] = 0.0f; }

  __syncthreads();

  const bf16* qrow = &Qs[(wave * 16 + lane16) * LQ];
  bf16* pbase = &Ps[wave * 16 * LP];

  for (int k0 = 0; k0 < SEQK; k0 += 64) {
    // V (manual, transposed scatter); K via TDM or manual
#pragma unroll
    for (int j = 0; j < 4; ++j) {
      int c = tid + 256 * j;
      int r = c >> 4;
      int cb = (c & 15) * 8;
      long long g = ((long long)(b * SEQK + k0 + r)) * HIDDEN + h * HEAD_DIM + cb;
      union { v4u v; bf16 e[8]; } u;
      u.v = *(const v4u*)&Vg[g];
#pragma unroll
      for (int e = 0; e < 8; ++e) Vst[(cb + e) * LV + r] = u.e[e];
#if !USE_TDM
      *(v4u*)&Ks[r * LK + cb] = *(const v4u*)&Kg[g];
#endif
    }
#if USE_TDM
    if (tid < 32) {
      tdm_load_2d(lds_off(Ks),
                  &Kg[((long long)(b * SEQK + k0)) * HIDDEN + h * HEAD_DIM],
                  128, 64, HIDDEN, 5, 3);
      __builtin_amdgcn_s_wait_tensorcnt(0);
    }
#endif
    __syncthreads();

    // S(16x64) = Q(16x128) * K^T via WMMA (K-dim = d)
    v8f S[4];
#pragma unroll
    for (int ct = 0; ct < 4; ++ct) S[ct] = zero8();
#pragma unroll
    for (int kk = 0; kk < 4; ++kk) {
      v16bf a = ldA(qrow, kk * 32, laneHi);
#pragma unroll
      for (int ct = 0; ct < 4; ++ct) {
        v16bf bb = ld16(&Ks[(ct * 16 + lane16) * LK + kk * 32 + 16 * laneHi]);
        S[ct] = wmma_bf16(a, bb, S[ct]);
      }
    }

    // mask + scale, row max (a row's 16 cols live in one 16-lane half)
    float rmax[8];
#pragma unroll
    for (int r = 0; r < 8; ++r) rmax[r] = -1e30f;
#pragma unroll
    for (int ct = 0; ct < 4; ++ct) {
      int col = k0 + ct * 16 + lane16;
      bool mk = maskg[b * SEQK + col] != 0;
#pragma unroll
      for (int r = 0; r < 8; ++r) {
        float s = mk ? S[ct][r] * scale : -1e30f;
        S[ct][r] = s;
        rmax[r] = fmaxf(rmax[r], s);
      }
    }
#pragma unroll
    for (int off = 1; off < 16; off <<= 1)
#pragma unroll
      for (int r = 0; r < 8; ++r)
        rmax[r] = fmaxf(rmax[r], __shfl_xor(rmax[r], off, 32));

    float alpha[8], rsum[8];
#pragma unroll
    for (int r = 0; r < 8; ++r) {
      float mnew = fmaxf(mrun[r], rmax[r]);
      alpha[r] = __expf(mrun[r] - mnew);
      mrun[r] = mnew;
      rsum[r] = 0.0f;
    }
#pragma unroll
    for (int ct = 0; ct < 4; ++ct)
#pragma unroll
      for (int r = 0; r < 8; ++r) {
        float p = __expf(S[ct][r] - mrun[r]);
        S[ct][r] = p;
        rsum[r] += p;
      }
#pragma unroll
    for (int off = 1; off < 16; off <<= 1)
#pragma unroll
      for (int r = 0; r < 8; ++r) rsum[r] += __shfl_xor(rsum[r], off, 32);
#pragma unroll
    for (int r = 0; r < 8; ++r) lrun[r] = lrun[r] * alpha[r] + rsum[r];
#pragma unroll
    for (int nd = 0; nd < 8; ++nd)
#pragma unroll
      for (int r = 0; r < 8; ++r) Oacc[nd][r] *= alpha[r];

    // P: C-layout -> LDS -> A-layout (same-wave DS is in-order)
#pragma unroll
    for (int ct = 0; ct < 4; ++ct)
#pragma unroll
      for (int r = 0; r < 8; ++r)
        pbase[(laneHi * 8 + r) * LP + ct * 16 + lane16] = f2bf(S[ct][r]);

    // O(16x128) += P(16x64) * V(64x128) (K-dim = kv; Vst row-major in kv)
#pragma unroll
    for (int kc = 0; kc < 2; ++kc) {
      v16bf a = ldA(&pbase[lane16 * LP], kc * 32, laneHi);
#pragma unroll
      for (int nd = 0; nd < 8; ++nd) {
        v16bf bb = ld16(&Vst[(nd * 16 + lane16) * LV + kc * 32 + 16 * laneHi]);
        Oacc[nd] = wmma_bf16(a, bb, Oacc[nd]);
      }
    }
    __syncthreads();
  }

#pragma unroll
  for (int nd = 0; nd < 8; ++nd)
#pragma unroll
    for (int r = 0; r < 8; ++r) {
      int m = wave * 16 + laneHi * 8 + r;
      int d = nd * 16 + lane16;
      float v = Oacc[nd][r] / lrun[r];
      O[((long long)(b * SEQQ + q0 + m)) * HIDDEN + h * HEAD_DIM + d] = f2bf(v);
    }
}

// ---------------- host orchestration ----------------

extern "C" void kernel_launch(void* const* d_in, const int* in_sizes, int n_in,
                              void* d_out, int out_size, void* d_ws, size_t ws_size,
                              hipStream_t stream) {
  (void)in_sizes; (void)n_in; (void)out_size; (void)ws_size;
  const float* x   = (const float*)d_in[0];
  const float* enc = (const float*)d_in[1];
  const int*   msk = (const int*)d_in[2];
  const float* Wq  = (const float*)d_in[3];
  const float* Wk  = (const float*)d_in[4];
  const float* Wv  = (const float*)d_in[5];
  const float* Wo  = (const float*)d_in[6];
  float* out = (float*)d_out;

  const size_t ACT_ELEMS = (size_t)BATCH * SEQQ * HIDDEN;
  const size_t WGT_ELEMS = (size_t)HIDDEN * HIDDEN;
  const size_t ACT_B = ACT_ELEMS * sizeof(bf16);
  const size_t WGT_B = WGT_ELEMS * sizeof(bf16);

  char* ws = (char*)d_ws;
  size_t off = 0;
  auto take = [&](size_t bytes) {
    void* p = ws + off;
    off += (bytes + 255) & ~(size_t)255;
    return p;
  };
  bf16* xb   = (bf16*)take(ACT_B);
  bf16* encb = (bf16*)take(ACT_B);
  bf16* wqb  = (bf16*)take(WGT_B);
  bf16* wkb  = (bf16*)take(WGT_B);
  bf16* wvb  = (bf16*)take(WGT_B);
  bf16* wob  = (bf16*)take(WGT_B);
  bf16* qpre = (bf16*)take(ACT_B);
  bf16* kpre = (bf16*)take(ACT_B);
  bf16* qb   = (bf16*)take(ACT_B);
  bf16* kb   = (bf16*)take(ACT_B);
  bf16* vb   = (bf16*)take(ACT_B);
  bf16* ab   = (bf16*)take(ACT_B);

  const int M = BATCH * SEQQ;
  const int N = HIDDEN;
  const int K = HIDDEN;

  {
    int n4a = (int)(ACT_ELEMS / 4), n4w = (int)(WGT_ELEMS / 4);
    dim3 blk(256);
    k_f32_to_bf16<<<dim3((n4a + 255) / 256), blk, 0, stream>>>(x,   xb,   n4a);
    k_f32_to_bf16<<<dim3((n4a + 255) / 256), blk, 0, stream>>>(enc, encb, n4a);
    k_f32_to_bf16<<<dim3((n4w + 255) / 256), blk, 0, stream>>>(Wq,  wqb,  n4w);
    k_f32_to_bf16<<<dim3((n4w + 255) / 256), blk, 0, stream>>>(Wk,  wkb,  n4w);
    k_f32_to_bf16<<<dim3((n4w + 255) / 256), blk, 0, stream>>>(Wv,  wvb,  n4w);
    k_f32_to_bf16<<<dim3((n4w + 255) / 256), blk, 0, stream>>>(Wo,  wob,  n4w);
  }

  dim3 ggrid(N / 128, M / 128), gblk(256);
  k_gemm_nt<false><<<ggrid, gblk, 0, stream>>>(xb,   wqb, qpre, M, N, K);
  k_gemm_nt<false><<<ggrid, gblk, 0, stream>>>(encb, wkb, kpre, M, N, K);
  k_gemm_nt<false><<<ggrid, gblk, 0, stream>>>(encb, wvb, vb,   M, N, K);

  {
    int npairs = BATCH * SEQQ * HEADS * (HEAD_DIM / 2);
    dim3 blk(256), grd((npairs + 255) / 256);
    k_rope<<<grd, blk, 0, stream>>>(qpre, qb, npairs);
    k_rope<<<grd, blk, 0, stream>>>(kpre, kb, npairs);
  }

  {
    size_t smem = (size_t)(128 * 136 + 64 * 136 + 128 * 72 + 8 * 16 * 72) * sizeof(bf16);
    k_attn<<<dim3(SEQQ / 128, HEADS, BATCH), dim3(256), smem, stream>>>(qb, kb, vb, msk, ab);
  }

  k_gemm_nt<true><<<ggrid, gblk, 0, stream>>>(ab, wob, out, M, N, K);
}
